// ann1_poisson_snn2_29171417874458
// MI455X (gfx1250) — compile-verified, matrix operand order checked by reading, not run on hardware
//
#include <hip/hip_runtime.h>
#include <math.h>

#define B_DIM 64
#define T_DIM 512
#define K1    784
#define K1P   800      /* 784 padded to 25*32 */
#define N1    1000
#define N1P   1024
#define F_DIM 500
#define F_PAD 512      /* 500 padded to 16*32 */
#define N3    10
#define N3P   16

#define EM_C 0.7788007830714049f   /* exp(-1/4) */
#define ES_C 0.36787944117144233f  /* exp(-1)   */

typedef __attribute__((ext_vector_type(16))) _Float16 v16h;
typedef __attribute__((ext_vector_type(8)))  _Float16 v8h;
typedef __attribute__((ext_vector_type(8)))  float    v8f;

// ---------------------------------------------------------------------------
// Register-blocked WMMA GEMM, no LDS: operands pre-packed f16, K%32==0,
// N padded so loads need no guards. A: (B, T, Kp) k-contiguous. W: (Np, Kp).
// Each wave computes MT m-tiles x NT n-tiles of 16x16.
// Y[b, o, t] = sum_k A[b,t,k]*W[o,k] + bias[o]  (optional ReLU)
// ---------------------------------------------------------------------------
template<int MT, int NT, int WV>
__global__ __launch_bounds__(WV * 32)
void gemm_wmma_reg(const _Float16* __restrict__ X,
                   const _Float16* __restrict__ Wp,
                   const float* __restrict__ bias,
                   float* __restrict__ Y,
                   int Kp, int N_out, int do_relu)
{
    const int lane = threadIdx.x & 31;
    const int wv   = threadIdx.x >> 5;
    const int row  = lane & 15;
    const int koff = (lane >> 4) * 8;   // lanes 0-15: K 0..7/16..23 ; 16-31: 8..15/24..31

    const int mpb = T_DIM / (16 * MT);              // m-blocks per batch image
    const int b   = blockIdx.x / mpb;
    const int t0  = (blockIdx.x % mpb) * (16 * MT);
    const int nb  = (blockIdx.y * WV + wv) * (16 * NT);

    const _Float16* Xb = X + ((size_t)b * T_DIM + t0) * Kp;

    v8f acc[MT][NT] = {};

    for (int k0 = 0; k0 < Kp; k0 += 32) {
        v16h af[MT], bfr[NT];
        #pragma unroll
        for (int mi = 0; mi < MT; ++mi) {
            const _Float16* p = Xb + (size_t)(mi * 16 + row) * Kp + k0 + koff;
            v8h lo = *(const v8h*)(p);
            v8h hi = *(const v8h*)(p + 16);
            #pragma unroll
            for (int j = 0; j < 8; ++j) { af[mi][j] = lo[j]; af[mi][j + 8] = hi[j]; }
        }
        #pragma unroll
        for (int ni = 0; ni < NT; ++ni) {
            const _Float16* p = Wp + (size_t)(nb + ni * 16 + row) * Kp + k0 + koff;
            v8h lo = *(const v8h*)(p);
            v8h hi = *(const v8h*)(p + 16);
            #pragma unroll
            for (int j = 0; j < 8; ++j) { bfr[ni][j] = lo[j]; bfr[ni][j + 8] = hi[j]; }
        }
        #pragma unroll
        for (int mi = 0; mi < MT; ++mi)
            #pragma unroll
            for (int ni = 0; ni < NT; ++ni)
                acc[mi][ni] = __builtin_amdgcn_wmma_f32_16x16x32_f16(
                    false, af[mi], false, bfr[ni], (short)0, acc[mi][ni], false, false);
    }

    // C/D layout: VGPR r -> M = r + 8*(lane>=16), N = lane&15
    const int n  = lane & 15;
    const int mo = (lane >> 4) * 8;
    #pragma unroll
    for (int mi = 0; mi < MT; ++mi)
        #pragma unroll
        for (int ni = 0; ni < NT; ++ni) {
            int o = nb + ni * 16 + n;
            if (o < N_out) {
                float bv = bias[o];
                float* Yp = Y + ((size_t)b * N_out + o) * T_DIM + t0 + mi * 16 + mo;
                #pragma unroll
                for (int r = 0; r < 8; ++r) {
                    float v = acc[mi][ni][r] + bv;
                    if (do_relu) v = fmaxf(v, 0.0f);
                    Yp[r] = v;
                }
            }
        }
}

// ---------------------------------------------------------------------------
// Pack inputs (B, 784, 512) f32 -> (B, 512, 800) f16, transposed + zero-pad k.
// 32x32 LDS tile transpose, both sides coalesced.
// ---------------------------------------------------------------------------
__global__ void pack_x_kernel(const float* __restrict__ in, _Float16* __restrict__ outp)
{
    __shared__ float tile[32][33];
    const int b  = blockIdx.z;
    const int k0 = blockIdx.x * 32;
    const int t0 = blockIdx.y * 32;
    for (int e = threadIdx.x; e < 1024; e += blockDim.x) {
        int kk = e >> 5, tt = e & 31;
        int k = k0 + kk;
        tile[kk][tt] = (k < K1) ? in[((size_t)b * K1 + k) * T_DIM + t0 + tt] : 0.0f;
    }
    __syncthreads();
    for (int e = threadIdx.x; e < 1024; e += blockDim.x) {
        int kk = e & 31, tt = e >> 5;
        outp[((size_t)b * T_DIM + t0 + tt) * K1P + k0 + kk] = (_Float16)tile[kk][tt];
    }
}

// Pack weights (N, K) f32 -> (Np, Kp) f16 zero-padded.
__global__ void pack_w_kernel(const float* __restrict__ w, _Float16* __restrict__ wp,
                              int N, int K, int Np, int Kp)
{
    int idx = blockIdx.x * blockDim.x + threadIdx.x;
    if (idx >= Np * Kp) return;
    int o = idx / Kp, k = idx - o * Kp;
    wp[idx] = (_Float16)((o < N && k < K) ? w[(size_t)o * K + k] : 0.0f);
}

// ---------------------------------------------------------------------------
// sample = mu + eps*exp(0.5*ln_var), dual-exp IIR along t; emits the padded
// f16 A-matrix (b, t, F_PAD) for GEMM2. One thread per (b, f) row; lanes cover
// consecutive f so the strided f16 stores coalesce across the wave.
// ---------------------------------------------------------------------------
__global__ void sample_iir_kernel(const float* __restrict__ ann,
                                  const float* __restrict__ eps,
                                  _Float16* __restrict__ y)
{
    int rowp = blockIdx.x * blockDim.x + threadIdx.x;
    if (rowp >= B_DIM * F_PAD) return;
    int b = rowp / F_PAD, f = rowp - b * F_PAD;
    _Float16* yp = y + (size_t)b * T_DIM * F_PAD + f;
    if (f >= F_DIM) {   // zero pad rows so GEMM2 needs no k-guard
        for (int t = 0; t < T_DIM; ++t) yp[(size_t)t * F_PAD] = (_Float16)0.0f;
        return;
    }
    const float4* mu4 = (const float4*)(ann + ((size_t)b * N1 + f) * T_DIM);
    const float4* lv4 = (const float4*)(ann + ((size_t)b * N1 + f + F_DIM) * T_DIM);
    const float4* e4  = (const float4*)(eps + ((size_t)b * F_DIM + f) * T_DIM);
    float m = 0.0f, s = 0.0f;
    for (int q = 0; q < T_DIM / 4; ++q) {
        float4 mu = mu4[q], lv = lv4[q], e = e4[q];
        float o[4], x;
        x = mu.x + e.x * __expf(0.5f * lv.x); m = EM_C*m + x; s = ES_C*s + x; o[0] = m - s;
        x = mu.y + e.y * __expf(0.5f * lv.y); m = EM_C*m + x; s = ES_C*s + x; o[1] = m - s;
        x = mu.z + e.z * __expf(0.5f * lv.z); m = EM_C*m + x; s = ES_C*s + x; o[2] = m - s;
        x = mu.w + e.w * __expf(0.5f * lv.w); m = EM_C*m + x; s = ES_C*s + x; o[3] = m - s;
        #pragma unroll
        for (int i = 0; i < 4; ++i) yp[(size_t)(4 * q + i) * F_PAD] = (_Float16)o[i];
    }
}

// Plain dual-exp IIR on spikes s2 (b,500,t) f32 -> padded f16 (b, t, F_PAD).
__global__ void iir_tr_kernel(const float* __restrict__ xin, _Float16* __restrict__ y)
{
    int rowp = blockIdx.x * blockDim.x + threadIdx.x;
    if (rowp >= B_DIM * F_PAD) return;
    int b = rowp / F_PAD, f = rowp - b * F_PAD;
    _Float16* yp = y + (size_t)b * T_DIM * F_PAD + f;
    if (f >= F_DIM) {
        for (int t = 0; t < T_DIM; ++t) yp[(size_t)t * F_PAD] = (_Float16)0.0f;
        return;
    }
    const float4* x4 = (const float4*)(xin + ((size_t)b * F_DIM + f) * T_DIM);
    float m = 0.0f, s = 0.0f;
    for (int q = 0; q < T_DIM / 4; ++q) {
        float4 x = x4[q];
        float o[4];
        m = EM_C*m + x.x; s = ES_C*s + x.x; o[0] = m - s;
        m = EM_C*m + x.y; s = ES_C*s + x.y; o[1] = m - s;
        m = EM_C*m + x.z; s = ES_C*s + x.z; o[2] = m - s;
        m = EM_C*m + x.w; s = ES_C*s + x.w; o[3] = m - s;
        #pragma unroll
        for (int i = 0; i < 4; ++i) yp[(size_t)(4 * q + i) * F_PAD] = (_Float16)o[i];
    }
}

// LIF: v = decay*v*(1-sp) + cur ; sp = (v >= 1). One thread per (b, o) row.
__global__ void lif_kernel(const float* __restrict__ wx,
                           float* __restrict__ out, int rows)
{
    int row = blockIdx.x * blockDim.x + threadIdx.x;
    if (row >= rows) return;
    const float4* c4 = (const float4*)(wx + (size_t)row * T_DIM);
    float4*       o4 = (float4*)(out + (size_t)row * T_DIM);
    float v = 0.0f, sp = 0.0f;
    for (int q = 0; q < T_DIM / 4; ++q) {
        float4 c = c4[q], o;
        v = EM_C*v*(1.0f - sp) + c.x; sp = (v >= 1.0f) ? 1.0f : 0.0f; o.x = sp;
        v = EM_C*v*(1.0f - sp) + c.y; sp = (v >= 1.0f) ? 1.0f : 0.0f; o.y = sp;
        v = EM_C*v*(1.0f - sp) + c.z; sp = (v >= 1.0f) ? 1.0f : 0.0f; o.z = sp;
        v = EM_C*v*(1.0f - sp) + c.w; sp = (v >= 1.0f) ? 1.0f : 0.0f; o.w = sp;
        o4[q] = o;
    }
}

extern "C" void kernel_launch(void* const* d_in, const int* in_sizes, int n_in,
                              void* d_out, int out_size, void* d_ws, size_t ws_size,
                              hipStream_t stream) {
    const float* inputs = (const float*)d_in[0];   // (64, 784, 512)
    const float* eps    = (const float*)d_in[1];   // (64, 500, 512)
    const float* w1     = (const float*)d_in[2];   // (1000, 784)
    const float* b1     = (const float*)d_in[3];
    const float* w2     = (const float*)d_in[4];   // (500, 500)
    const float* b2     = (const float*)d_in[5];
    const float* w3     = (const float*)d_in[6];   // (10, 500)
    const float* b3     = (const float*)d_in[7];
    float* out = (float*)d_out;                    // (64, 10, 512)

    // ---- workspace layout (ping-pong) ----
    _Float16* X1p = (_Float16*)d_ws;                       // 64*512*800 halfs
    _Float16* W1p = X1p + (size_t)B_DIM * T_DIM * K1P;     // 1024*800
    _Float16* W2p = W1p + (size_t)N1P * K1P;               // 512*512
    _Float16* W3p = W2p + (size_t)F_PAD * F_PAD;           // 16*512
    float*    ann = (float*)(W3p + (size_t)N3P * F_PAD);   // 64*1000*512 f32
    float*    wx2 = ann;                                   // reuse after ann dead
    float*    s2  = ann + (size_t)B_DIM * F_DIM * T_DIM;   // second half of region
    _Float16* a2p = X1p;                                   // reuse after GEMM1
    _Float16* a3p = (_Float16*)wx2;                        // reuse after LIF2
    float*    wx3 = (float*)X1p;                           // reuse after GEMM2

    // ---- 0) pack operands to padded f16 ----
    pack_x_kernel<<<dim3(K1P / 32, T_DIM / 32, B_DIM), 256, 0, stream>>>(inputs, X1p);
    pack_w_kernel<<<(N1P * K1P + 255) / 256, 256, 0, stream>>>(w1, W1p, N1, K1, N1P, K1P);
    pack_w_kernel<<<(F_PAD * F_PAD + 255) / 256, 256, 0, stream>>>(w2, W2p, F_DIM, F_DIM, F_PAD, F_PAD);
    pack_w_kernel<<<(N3P * F_PAD + 255) / 256, 256, 0, stream>>>(w3, W3p, N3, F_DIM, N3P, F_PAD);

    // ---- 1) ann = relu(inputs @ w1^T + b1) -> (B,1000,T) f32 ----
    gemm_wmma_reg<2, 2, 4><<<dim3(B_DIM * (T_DIM / 32), N1P / 128), 128, 0, stream>>>(
        X1p, W1p, b1, ann, K1P, N1, 1);

    // ---- 2) reparam sample + IIR -> a2 packed f16 (B, T, 512) ----
    sample_iir_kernel<<<(B_DIM * F_PAD + 255) / 256, 256, 0, stream>>>(ann, eps, a2p);

    // ---- 3) wx2 = a2 @ w2^T + b2 -> (B,500,T) f32 ----
    gemm_wmma_reg<2, 2, 4><<<dim3(B_DIM * (T_DIM / 32), F_PAD / 128), 128, 0, stream>>>(
        a2p, W2p, b2, wx2, F_PAD, F_DIM, 0);

    // ---- 4) LIF -> s2 (B,500,T) f32 ----
    lif_kernel<<<(B_DIM * F_DIM + 255) / 256, 256, 0, stream>>>(wx2, s2, B_DIM * F_DIM);

    // ---- 5) IIR -> a3 packed f16 (B, T, 512) ----
    iir_tr_kernel<<<(B_DIM * F_PAD + 255) / 256, 256, 0, stream>>>(s2, a3p);

    // ---- 6) wx3 = a3 @ w3^T + b3 -> (B,10,T) f32 ----
    gemm_wmma_reg<2, 1, 1><<<dim3(B_DIM * (T_DIM / 32), 1), 32, 0, stream>>>(
        a3p, W3p, b3, wx3, F_PAD, N3, 0);

    // ---- 7) final LIF -> d_out (B,10,T) ----
    lif_kernel<<<(B_DIM * N3 + 255) / 256, 256, 0, stream>>>(wx3, out, B_DIM * N3);
}
